// MemoryEfficientAttention_50964081934771
// MI455X (gfx1250) — matrix operand order repaired and checked by reading, not compiled, not run on hardware
//
#include <hip/hip_runtime.h>

typedef __bf16 bf16;
typedef __attribute__((ext_vector_type(16))) bf16 v16bf;
typedef __attribute__((ext_vector_type(8)))  bf16 v8bf;
typedef __attribute__((ext_vector_type(8)))  float v8f;

#define WMMA_BF16(a,b,c) __builtin_amdgcn_wmma_f32_16x16x32_bf16(false,(a),false,(b),(short)0,(c),false,false)

// Problem sizes (fixed by the reference)
constexpr int Bv = 2, Sv = 2048, Hv = 2048, NHv = 16, HDv = 128;
constexpr int Mv = Bv * Sv;   // 4096 rows of x
constexpr int Kv = Hv;        // 2048 reduction dim
constexpr int Nv = Hv;        // 2048 output cols

// ---------------------------------------------------------------------------
// fp32 -> bf16 conversion (one-time, bandwidth-trivial)
// ---------------------------------------------------------------------------
__global__ void cvt_f32_bf16(const float* __restrict__ in, bf16* __restrict__ out, int n) {
    int i = blockIdx.x * blockDim.x + threadIdx.x;
    int stride = gridDim.x * blockDim.x;
    for (; i < n; i += stride) out[i] = (bf16)in[i];
}

// ---------------------------------------------------------------------------
// WMMA GEMM: C[M,N] = A[M,K] @ W[N,K]^T   (nn.Linear convention)
// mode 0: fp32 out + bias               -> Cf[M,N]
// mode 1: bf16 out, head-split          -> Cb[B,NH,S,HD]   (Q, K)
// mode 2: bf16 out, head-split + transp -> Cb[B,NH,HD,S]   (V^T)
//
// Wave computes 64x64 (4x4 WMMA frags): each A fragment is reused across 4 B
// fragments and vice versa -> 16 WMMAs per K=32 step against 8 KB of fragment
// traffic = 32 flops/byte from L2 (2x the 32x32 tile). 4 waves (2x2) per
// block -> 128x128 block tile. All fragment loads are contiguous b128s
// (A row-striped; B frag lane n holds W row n, contiguous along K).
// ---------------------------------------------------------------------------
__global__ __launch_bounds__(128)
void gemm_bf16(const bf16* __restrict__ A, const bf16* __restrict__ Wm,
               const float* __restrict__ bias, float* __restrict__ Cf,
               bf16* __restrict__ Cb, int mode) {
    const int lane = threadIdx.x & 31;
    const int w    = threadIdx.x >> 5;     // 0..3
    const int hi   = lane >> 4;            // 0/1 half-wave
    const int ln   = lane & 15;
    const int m0   = blockIdx.y * 128 + (w >> 1) * 64;
    const int n0   = blockIdx.x * 128 + (w & 1) * 64;

    v8f c[4][4];
    #pragma unroll
    for (int mi = 0; mi < 4; mi++)
        #pragma unroll
        for (int ni = 0; ni < 4; ni++)
            c[mi][ni] = (v8f){0,0,0,0,0,0,0,0};

    for (int kb = 0; kb < Kv; kb += 32) {
        v16bf a[4], b[4];
        // A-frags: lanes 0-15 take K 0..15, lanes 16-31 K 16..31 (contiguous)
        const int ca = kb + (hi ? 16 : 0);
        #pragma unroll
        for (int mi = 0; mi < 4; mi++) {
            const bf16* p = A + (size_t)(m0 + mi * 16 + ln) * Kv + ca;
            v8bf lo = *(const v8bf*)p, hh = *(const v8bf*)(p + 8);
            #pragma unroll
            for (int i = 0; i < 8; i++) { a[mi][i] = lo[i]; a[mi][i+8] = hh[i]; }
        }
        // B-frags: lane n = output col = W row n; K contiguous (hi half K+16)
        const int cb = kb + (hi ? 16 : 0);
        #pragma unroll
        for (int ni = 0; ni < 4; ni++)
            b[ni] = *(const v16bf*)(Wm + (size_t)(n0 + ni * 16 + ln) * Kv + cb);

        #pragma unroll
        for (int mi = 0; mi < 4; mi++)
            #pragma unroll
            for (int ni = 0; ni < 4; ni++)
                c[mi][ni] = WMMA_BF16(a[mi], b[ni], c[mi][ni]);
    }

    // Epilogue: C layout row = v + 8*hi, col = ln (per 16x16 frag)
    #pragma unroll
    for (int mi = 0; mi < 4; mi++) {
        #pragma unroll
        for (int ni = 0; ni < 4; ni++) {
            const int n = n0 + ni * 16 + ln;
            #pragma unroll
            for (int v = 0; v < 8; v++) {
                const int r = m0 + mi * 16 + v + 8 * hi;
                const float val = c[mi][ni][v];
                if (mode == 0) {
                    Cf[(size_t)r * Nv + n] = val + bias[n];
                } else {
                    const int bidx = r >> 11, s = r & (Sv - 1);     // Sv=2048
                    const int head = n >> 7,  hd = n & (HDv - 1);   // HDv=128
                    if (mode == 1)
                        Cb[(((size_t)(bidx * NHv + head)) * Sv + s) * HDv + hd] = (bf16)val;
                    else
                        Cb[(((size_t)(bidx * NHv + head)) * HDv + hd) * Sv + s] = (bf16)val;
                }
            }
        }
    }
}

// ---------------------------------------------------------------------------
// Flash attention: one block per (b*NH, 64 q-rows); 4 waves x 16 q-rows each.
// Q: [BH,S,HD] bf16, K: [BH,S,HD] bf16, Vt: [BH,HD,S] bf16 (transposed so PV
// B-fragments are contiguous along the key dim). Out: bf16 [B,S,H].
// Online softmax; row stats live per-lane matching the WMMA C layout
// (lanes 0-15 own rows v, lanes 16-31 rows v+8; shfl_xor masks 1..8 reduce
// within each 16-lane half).
// ---------------------------------------------------------------------------
__global__ __launch_bounds__(128)
void flash_attn(const bf16* __restrict__ Q, const bf16* __restrict__ Kc,
                const bf16* __restrict__ Vt, bf16* __restrict__ Out) {
    __shared__ __align__(16) bf16 plds[4][16][32];
    const int lane = threadIdx.x & 31;
    const int w    = threadIdx.x >> 5;   // 0..3
    const int hi   = lane >> 4;
    const int ln   = lane & 15;
    const int bh   = blockIdx.x;                 // 0..31
    const int q0   = blockIdx.y * 64 + w * 16;   // q-row base in S
    const float scale = 0.088388347648318447f;   // 1/sqrt(128)

    const bf16* Qb = Q  + (size_t)bh * Sv * HDv;
    const bf16* Kb = Kc + (size_t)bh * Sv * HDv;
    const bf16* Vb = Vt + (size_t)bh * HDv * Sv;

    // Q tile 16x128 as 4 A-frags, kept resident
    v16bf qf[4];
    #pragma unroll
    for (int kk = 0; kk < 4; kk++) {
        const int c0 = kk * 32 + (hi ? 16 : 0);
        const bf16* p = Qb + (size_t)(q0 + ln) * HDv + c0;
        v8bf lo = *(const v8bf*)p, hh = *(const v8bf*)(p + 8);
        #pragma unroll
        for (int i = 0; i < 8; i++) { qf[kk][i] = lo[i]; qf[kk][i+8] = hh[i]; }
    }

    v8f zero = {0,0,0,0,0,0,0,0};
    v8f o[8];
    #pragma unroll
    for (int i = 0; i < 8; i++) o[i] = zero;
    float m[8], l[8];
    #pragma unroll
    for (int v = 0; v < 8; v++) { m[v] = -3.0e38f; l[v] = 0.0f; }

    for (int j = 0; j < Sv; j += 32) {
        // scores 16x32 = Q(16x128) @ K_j(32x128)^T : 8 WMMAs
        v8f s0 = zero, s1 = zero;
        #pragma unroll
        for (int kk = 0; kk < 4; kk++) {
            const int cb = kk * 32 + (hi ? 16 : 0);
            v16bf kf0 = *(const v16bf*)(Kb + (size_t)(j + ln) * HDv + cb);
            v16bf kf1 = *(const v16bf*)(Kb + (size_t)(j + 16 + ln) * HDv + cb);
            s0 = WMMA_BF16(qf[kk], kf0, s0);
            s1 = WMMA_BF16(qf[kk], kf1, s1);
        }

        // online softmax update (per-lane rows match C layout)
        float p0[8], p1[8];
        #pragma unroll
        for (int v = 0; v < 8; v++) {
            p0[v] = s0[v] * scale;
            p1[v] = s1[v] * scale;
            float t = fmaxf(p0[v], p1[v]);
            t = fmaxf(t, __shfl_xor(t, 1, 32));
            t = fmaxf(t, __shfl_xor(t, 2, 32));
            t = fmaxf(t, __shfl_xor(t, 4, 32));
            t = fmaxf(t, __shfl_xor(t, 8, 32));
            const float mn = fmaxf(m[v], t);
            const float al = __expf(m[v] - mn);
            m[v] = mn;
            p0[v] = __expf(p0[v] - mn);
            p1[v] = __expf(p1[v] - mn);
            float sm = p0[v] + p1[v];
            sm += __shfl_xor(sm, 1, 32);
            sm += __shfl_xor(sm, 2, 32);
            sm += __shfl_xor(sm, 4, 32);
            sm += __shfl_xor(sm, 8, 32);
            l[v] = l[v] * al + sm;
            #pragma unroll
            for (int nt = 0; nt < 8; nt++) o[nt][v] *= al;
        }

        // C-layout -> A-layout for P via LDS (per-wave 16x32 bf16 region)
        #pragma unroll
        for (int v = 0; v < 8; v++) {
            const int r = v + 8 * hi;
            plds[w][r][ln]      = (bf16)p0[v];
            plds[w][r][ln + 16] = (bf16)p1[v];
        }
        __syncthreads();
        v16bf pa;
        {
            const int c0 = hi ? 16 : 0;
            const bf16* p = &plds[w][ln][c0];
            v8bf lo = *(const v8bf*)p, hh = *(const v8bf*)(p + 8);
            #pragma unroll
            for (int i = 0; i < 8; i++) { pa[i] = lo[i]; pa[i+8] = hh[i]; }
        }

        // O(16x128) += P(16x32) @ V_j(32x128) : 8 WMMAs (Vt rows contiguous in keys)
        #pragma unroll
        for (int nt = 0; nt < 8; nt++) {
            v16bf vf = *(const v16bf*)(Vb + (size_t)(nt * 16 + ln) * Sv + j + (hi ? 16 : 0));
            o[nt] = WMMA_BF16(pa, vf, o[nt]);
        }
        __syncthreads();
    }

    // normalize and write bf16 attention output in merged-head [B,S,H] layout
    const int b = bh / NHv, h = bh % NHv;
    #pragma unroll
    for (int v = 0; v < 8; v++) {
        const float inv = 1.0f / l[v];
        const int s = q0 + v + 8 * hi;
        #pragma unroll
        for (int nt = 0; nt < 8; nt++) {
            const int col = h * HDv + nt * 16 + ln;
            Out[((size_t)(b * Sv + s)) * Hv + col] = (bf16)(o[nt][v] * inv);
        }
    }
}

// ---------------------------------------------------------------------------
extern "C" void kernel_launch(void* const* d_in, const int* in_sizes, int n_in,
                              void* d_out, int out_size, void* d_ws, size_t ws_size,
                              hipStream_t stream) {
    (void)in_sizes; (void)n_in; (void)out_size; (void)ws_size;
    const float* x  = (const float*)d_in[0];
    const float* Wq = (const float*)d_in[1];
    const float* Wk = (const float*)d_in[2];
    const float* Wv = (const float*)d_in[3];
    const float* Wo = (const float*)d_in[4];
    const float* bo = (const float*)d_in[5];
    float* out = (float*)d_out;

    // Workspace carve-up (total 112 MB)
    char* ws = (char*)d_ws;
    bf16* xb   = (bf16*)ws; ws += (size_t)Mv * Kv * 2;  // 16 MB
    bf16* wqb  = (bf16*)ws; ws += (size_t)Hv * Hv * 2;  //  8 MB
    bf16* wkb  = (bf16*)ws; ws += (size_t)Hv * Hv * 2;
    bf16* wvb  = (bf16*)ws; ws += (size_t)Hv * Hv * 2;
    bf16* wob  = (bf16*)ws; ws += (size_t)Hv * Hv * 2;
    bf16* qb   = (bf16*)ws; ws += (size_t)Mv * Hv * 2;  // [BH,S,HD]
    bf16* kb   = (bf16*)ws; ws += (size_t)Mv * Hv * 2;  // [BH,S,HD]
    bf16* vtb  = (bf16*)ws; ws += (size_t)Mv * Hv * 2;  // [BH,HD,S]
    bf16* attb = (bf16*)ws; ws += (size_t)Mv * Hv * 2;  // [B,S,H]

    // fp32 -> bf16
    cvt_f32_bf16<<<dim3(2048), dim3(256), 0, stream>>>(x,  xb,  Mv * Kv);
    cvt_f32_bf16<<<dim3(1024), dim3(256), 0, stream>>>(Wq, wqb, Hv * Hv);
    cvt_f32_bf16<<<dim3(1024), dim3(256), 0, stream>>>(Wk, wkb, Hv * Hv);
    cvt_f32_bf16<<<dim3(1024), dim3(256), 0, stream>>>(Wv, wvb, Hv * Hv);
    cvt_f32_bf16<<<dim3(1024), dim3(256), 0, stream>>>(Wo, wob, Hv * Hv);

    // QKV projections (Q/K head-split, V head-split + transposed)
    dim3 gg(Nv / 128, Mv / 128);
    gemm_bf16<<<gg, 128, 0, stream>>>(xb, wqb, nullptr, nullptr, qb,  1);
    gemm_bf16<<<gg, 128, 0, stream>>>(xb, wkb, nullptr, nullptr, kb,  1);
    gemm_bf16<<<gg, 128, 0, stream>>>(xb, wvb, nullptr, nullptr, vtb, 2);

    // Flash attention: (b*NH) x (S/64) blocks, 128 threads
    flash_attn<<<dim3(Bv * NHv, Sv / 64), 128, 0, stream>>>(qb, kb, vtb, attb);

    // Output projection + bias -> fp32 d_out
    gemm_bf16<<<gg, 128, 0, stream>>>(attb, wob, bo, out, nullptr, 0);
}
